// MambaIndexer_68985764708574
// MI455X (gfx1250) — matrix operand order repaired and checked by reading, not compiled, not run on hardware
//
#include <hip/hip_runtime.h>
#include <hip/hip_bf16.h>
#include <math.h>

// Problem dimensions (compile-time constants from the reference)
#define LQ      4096   // SEQ
#define DM      1024   // D_MODEL
#define DIP     2208   // D_IN_PROJ
#define DIN     1024   // D_INNER
#define CONVD   1152   // D_INNER + 2*D_STATE
#define DSTATE  64
#define NH      32
#define HD      32
#define IDXD    64
#define DTOFF   2176   // D_INNER + CONV_DIM
#define TC      64     // SSD chunk length
#define NC      (LQ / TC)

typedef __attribute__((ext_vector_type(16))) __bf16 v16bf;
typedef __attribute__((ext_vector_type(8)))  __bf16 v8bf;
typedef __attribute__((ext_vector_type(4)))  __bf16 v4bf;
typedef __attribute__((ext_vector_type(8)))  float  v8f;
typedef __attribute__((ext_vector_type(4)))  float  v4f;

// ---------------------------------------------------------------------------
// WMMA fragment loaders (wave32, V_WMMA_F32_16X16X32_BF16 layouts, ISA 7.12.2)
// ---------------------------------------------------------------------------
__device__ __forceinline__ v16bf load_a_frag(const __bf16* base, int ld, int lane) {
  int row = lane & 15;
  int kh  = (lane >> 4) & 1;
  const __bf16* p = base + (size_t)row * ld + kh * 8;
  v8bf lo = *(const v8bf*)(p);
  v8bf hi = *(const v8bf*)(p + 16);
  return __builtin_shufflevector(lo, hi, 0,1,2,3,4,5,6,7,8,9,10,11,12,13,14,15);
}

__device__ __forceinline__ v16bf load_b_frag(const __bf16* base, int ld, int lane) {
  int col = lane & 15;
  int kh  = (lane >> 4) & 1;
  const __bf16* p = base + (size_t)col * ld + kh * 8;
  v8bf lo = *(const v8bf*)(p);
  v8bf hi = *(const v8bf*)(p + 16);
  return __builtin_shufflevector(lo, hi, 0,1,2,3,4,5,6,7,8,9,10,11,12,13,14,15);
}

__device__ __forceinline__ v8f wmma_bf16(v16bf a, v16bf b, v8f c) {
  return __builtin_amdgcn_wmma_f32_16x16x32_bf16(
      false, a, false, b, (short)0, c, false, false);
}

// ---------------------------------------------------------------------------
// fp32 -> bf16 conversion (vectorized x4)
// ---------------------------------------------------------------------------
__global__ void k_f32_to_bf16(const float* __restrict__ in, __bf16* __restrict__ out, int n4) {
  int i = blockIdx.x * blockDim.x + threadIdx.x;
  if (i >= n4) return;
  v4f f = *(const v4f*)(in + i * 4);
  v4bf o;
#pragma unroll
  for (int u = 0; u < 4; ++u) o[u] = (__bf16)f[u];
  *(v4bf*)(out + i * 4) = o;
}

// ---------------------------------------------------------------------------
// WMMA GEMM: out[M,N] = A[M,K] @ W[N,K]^T. One 16x64 strip per wave:
// A fragment loaded once per k-step and reused across 4 B tiles.
// ---------------------------------------------------------------------------
template<bool OUT_BF16>
__global__ void k_gemm_wmma(const __bf16* __restrict__ A,
                            const __bf16* __restrict__ W,
                            float*  __restrict__ outF,
                            __bf16* __restrict__ outB,
                            int M, int N, int K) {
  int lane = threadIdx.x & 31;
  int wave = threadIdx.x >> 5;
  int ntiles = N >> 4;
  int nt0 = (blockIdx.x * 4 + wave) * 4;
  int mt  = blockIdx.y;
  if (nt0 >= ntiles) return;

  const __bf16* Abase = A + (size_t)mt * 16 * K;
  v8f acc[4] = {{}, {}, {}, {}};

  for (int k0 = 0; k0 < K; k0 += 32) {
    if (k0 + 64 < K) __builtin_prefetch(Abase + k0 + 64, 0, 3);
    v16bf a = load_a_frag(Abase + k0, K, lane);
#pragma unroll
    for (int j = 0; j < 4; ++j) {
      // overreads past N stay inside the workspace; stores are guarded
      v16bf b = load_b_frag(W + (size_t)(nt0 + j) * 16 * K + k0, K, lane);
      acc[j] = wmma_bf16(a, b, acc[j]);
    }
  }

  int r0 = mt * 16 + ((lane >> 4) << 3);
#pragma unroll
  for (int j = 0; j < 4; ++j) {
    if (nt0 + j >= ntiles) break;
    int c = (nt0 + j) * 16 + (lane & 15);
    if (OUT_BF16) {
#pragma unroll
      for (int r = 0; r < 8; ++r) outB[(size_t)(r0 + r) * N + c] = (__bf16)acc[j][r];
    } else {
#pragma unroll
      for (int r = 0; r < 8; ++r) outF[(size_t)(r0 + r) * N + c] = acc[j][r];
    }
  }
}

// ---------------------------------------------------------------------------
// Depthwise causal conv1d (K=4) + bias + SiLU; 4 channels per thread (float4)
// ---------------------------------------------------------------------------
__global__ void k_conv_silu(const float* __restrict__ zx,
                            const float* __restrict__ cw,
                            const float* __restrict__ cb,
                            float* __restrict__ xc) {
  int idx = blockIdx.x * blockDim.x + threadIdx.x;   // over LQ * CONVD/4
  if (idx >= LQ * (CONVD / 4)) return;
  int t  = idx / (CONVD / 4);
  int c0 = (idx - t * (CONVD / 4)) * 4;

  v4f wv[4];                                         // wv[u] = cw[c0+u][0..3]
#pragma unroll
  for (int u = 0; u < 4; ++u) wv[u] = *(const v4f*)(cw + (c0 + u) * 4);

  v4f acc = *(const v4f*)(cb + c0);
#pragma unroll
  for (int j = 0; j < 4; ++j) {
    int tt = t - 3 + j;
    if (tt >= 0) {
      v4f xv = *(const v4f*)(zx + (size_t)tt * DIP + DIN + c0);
#pragma unroll
      for (int u = 0; u < 4; ++u) acc[u] += xv[u] * wv[u][j];
    }
  }
  v4f o;
#pragma unroll
  for (int u = 0; u < 4; ++u) o[u] = acc[u] / (1.f + __expf(-acc[u]));
  *(v4f*)(xc + (size_t)t * CONVD + c0) = o;
}

// ---------------------------------------------------------------------------
// dt = softplus(dt_raw + bias);  ldA = dt * (-exp(A_log))  (= log of dA)
// ---------------------------------------------------------------------------
__global__ void k_dt(const float* __restrict__ zx,
                     const float* __restrict__ dt_bias,
                     const float* __restrict__ A_log,
                     float* __restrict__ dtp,
                     float* __restrict__ ldA) {
  int idx = blockIdx.x * blockDim.x + threadIdx.x;
  if (idx >= LQ * NH) return;
  int t = idx / NH;
  int h = idx - t * NH;
  float d  = zx[(size_t)t * DIP + DTOFF + h] + dt_bias[h];
  float sp = (d > 20.f) ? d : log1pf(__expf(d));
  dtp[idx] = sp;
  ldA[idx] = sp * (-__expf(A_log[h]));
}

// ---------------------------------------------------------------------------
// SSD chunked scan, phase 1 (parallel over (chunk, head)):
//   CBt = B @ C^T                       (WMMA, 4x4 tiles, K=64)
//   G[t,s] = CBt[s,t]*exp(lp[t]-lp[s])*dt[s], masked s<=t (branchless,
//            vector ds_store_b128: lane's 8 accs run along s at fixed t)
//   Y_intra = G @ X^T                   (WMMA, 4x2 tiles)
//   U = (w*X)^T @ B                     (WMMA, 2x4 tiles)
// lp prefix computed with a parallel Hillis-Steele scan.
// ---------------------------------------------------------------------------
__global__ void k_ssd_chunk(const float* __restrict__ xc,
                            const float* __restrict__ dtp,
                            const float* __restrict__ ldA,
                            float* __restrict__ ysv,    // [LQ,DIN]  Y_intra
                            float* __restrict__ Uglob,  // [NC*NH][HD*64]
                            float* __restrict__ Pglob,  // [LQ*NH]
                            float* __restrict__ Ptot) { // [NC*NH]
  int c = blockIdx.x;
  int h = blockIdx.y;
  int T0 = c * TC;
  int tid = threadIdx.x;        // 128
  int lane = tid & 31;
  int wave = tid >> 5;

  __shared__ __align__(16) __bf16 sB [TC * 64];   // B[s][n]
  __shared__ __align__(16) __bf16 sBt[64 * TC];   // B^T[n][s]
  __shared__ __align__(16) __bf16 sC [TC * 64];   // C[t][n]
  __shared__ __align__(16) __bf16 sXt[HD * TC];   // X^T[p][s]
  __shared__ __align__(16) __bf16 sWx[HD * TC];   // (w*X)^T[p][s]
  __shared__ __align__(16) __bf16 sG [TC * TC];   // G[t][s]
  __shared__ float sdt[TC], slp[TC], sw[TC];

  // ---- stage B, Bt, C (float4 loads, v4bf stores) ----
  for (int i = tid; i < TC * 16; i += 128) {
    int s = i >> 4, n0 = (i & 15) * 4;
    const float* row = xc + (size_t)(T0 + s) * CONVD;
    v4f bv = *(const v4f*)(row + DIN + n0);
    v4f cv = *(const v4f*)(row + DIN + DSTATE + n0);
    v4bf bb, cc;
#pragma unroll
    for (int u = 0; u < 4; ++u) { bb[u] = (__bf16)bv[u]; cc[u] = (__bf16)cv[u]; }
    *(v4bf*)(sB + s * 64 + n0) = bb;
    *(v4bf*)(sC + s * 64 + n0) = cc;
#pragma unroll
    for (int u = 0; u < 4; ++u) sBt[(n0 + u) * TC + s] = bb[u];
  }
  // ---- stage X^T ----
  for (int i = tid; i < TC * 8; i += 128) {
    int s = i >> 3, p0 = (i & 7) * 4;
    v4f xv = *(const v4f*)(xc + (size_t)(T0 + s) * CONVD + h * HD + p0);
#pragma unroll
    for (int u = 0; u < 4; ++u) sXt[(p0 + u) * TC + s] = (__bf16)xv[u];
  }
  // ---- load per-step scalars ----
  if (tid < TC) {
    slp[tid] = ldA[(T0 + tid) * NH + h];
    sdt[tid] = dtp[(T0 + tid) * NH + h];
  }
  __syncthreads();

  // ---- parallel inclusive prefix sum of log(dA) ----
#pragma unroll
  for (int off = 1; off < TC; off <<= 1) {
    float add = 0.f;
    if (tid < TC && tid >= off) add = slp[tid - off];
    __syncthreads();
    if (tid < TC) slp[tid] += add;
    __syncthreads();
  }
  if (tid < TC) Pglob[(size_t)(T0 + tid) * NH + h] = __expf(slp[tid]);
  __syncthreads();
  float lptot = slp[TC - 1];
  if (tid < TC) sw[tid] = __expf(lptot - slp[tid]) * sdt[tid];
  if (tid == 0) Ptot[c * NH + h] = __expf(lptot);
  __syncthreads();

  // ---- stage (w*X)^T ----
  for (int i = tid; i < TC * 8; i += 128) {
    int s = i >> 3, p0 = (i & 7) * 4;
    v4f xv = *(const v4f*)(xc + (size_t)(T0 + s) * CONVD + h * HD + p0);
    float w = sw[s];
#pragma unroll
    for (int u = 0; u < 4; ++u) sWx[(p0 + u) * TC + s] = (__bf16)(w * xv[u]);
  }
  __syncthreads();

  // ---- CBt = B @ C^T, fold decay/mask, store G[t][s] with b128 stores ----
#pragma unroll
  for (int i = 0; i < 4; ++i) {
    int tile = wave * 4 + i;
    int tm = tile >> 2, tn = tile & 3;     // tm -> s tiles, tn -> t tiles
    v8f acc = {};
#pragma unroll
    for (int k0 = 0; k0 < 64; k0 += 32) {
      v16bf a = load_a_frag(sB + tm * 16 * 64 + k0, 64, lane);
      v16bf b = load_b_frag(sC + tn * 16 * 64 + k0, 64, lane);
      acc = wmma_bf16(a, b, acc);
    }
    int s0 = tm * 16 + ((lane >> 4) << 3);
    int t  = tn * 16 + (lane & 15);
    float lpt = slp[t];
    v8bf gv;
#pragma unroll
    for (int r = 0; r < 8; ++r) {
      int s = s0 + r;
      float g = __expf(lpt - slp[s]) * sdt[s];
      g = (s <= t) ? g : 0.f;              // branchless cndmask
      gv[r] = (__bf16)(g * acc[r]);
    }
    *(v8bf*)(sG + t * TC + s0) = gv;
  }
  __syncthreads();

  // ---- Y_intra = G @ X^T : 8 tiles (4 x 2) ----
#pragma unroll
  for (int i = 0; i < 2; ++i) {
    int tile = wave * 2 + i;
    int tm = tile >> 1, tn = tile & 1;
    v8f acc = {};
#pragma unroll
    for (int k0 = 0; k0 < TC; k0 += 32) {
      v16bf a = load_a_frag(sG + tm * 16 * TC + k0, TC, lane);
      v16bf b = load_b_frag(sXt + tn * 16 * TC + k0, TC, lane);
      acc = wmma_bf16(a, b, acc);
    }
    int r0 = tm * 16 + ((lane >> 4) << 3);
    int p  = tn * 16 + (lane & 15);
#pragma unroll
    for (int r = 0; r < 8; ++r)
      ysv[(size_t)(T0 + r0 + r) * DIN + h * HD + p] = acc[r];
  }

  // ---- U = (w*X)^T @ B : 8 tiles (2 x 4) ----
  float* Up = Uglob + (size_t)(c * NH + h) * (HD * 64);
#pragma unroll
  for (int i = 0; i < 2; ++i) {
    int tile = wave * 2 + i;
    int tm = tile >> 2, tn = tile & 3;
    v8f acc = {};
#pragma unroll
    for (int k0 = 0; k0 < TC; k0 += 32) {
      v16bf a = load_a_frag(sWx + tm * 16 * TC + k0, TC, lane);
      v16bf b = load_b_frag(sBt + tn * 16 * TC + k0, TC, lane);
      acc = wmma_bf16(a, b, acc);
    }
    int r0 = tm * 16 + ((lane >> 4) << 3);
    int n  = tn * 16 + (lane & 15);
#pragma unroll
    for (int r = 0; r < 8; ++r)
      Up[(r0 + r) * 64 + n] = acc[r];
  }
}

// ---------------------------------------------------------------------------
// SSD phase 2: cross-chunk state recurrence (tiny serial part).
// ---------------------------------------------------------------------------
__global__ void k_chunk_state(const float* __restrict__ Uglob,
                              const float* __restrict__ Ptot,
                              float* __restrict__ Hin) {
  int h = blockIdx.x;
  int tid = threadIdx.x;          // 1024; 2048 states per head
  int i0 = tid, i1 = tid + 1024;
  float st0 = 0.f, st1 = 0.f;
  for (int c = 0; c < NC; ++c) {
    size_t base = (size_t)(c * NH + h) * 2048;
    Hin[base + i0] = st0;
    Hin[base + i1] = st1;
    float pt = Ptot[c * NH + h];
    st0 = pt * st0 + Uglob[base + i0];
    st1 = pt * st1 + Uglob[base + i1];
  }
}

// ---------------------------------------------------------------------------
// SSD phase 3: Y_inter[t,p] = P[t] * (C[t,:] . Hin[p,:]) added into ysv.
// ---------------------------------------------------------------------------
__global__ void k_ssd_inter(const float* __restrict__ xc,
                            const float* __restrict__ Hin,
                            const float* __restrict__ Pglob,
                            float* __restrict__ ysv) {
  int c = blockIdx.x, h = blockIdx.y;
  int T0 = c * TC;
  int tid = threadIdx.x, lane = tid & 31, wave = tid >> 5;

  __shared__ __align__(16) __bf16 sC[TC * 64];
  __shared__ __align__(16) __bf16 sH[HD * 64];   // Hin[p][n]

  for (int i = tid; i < TC * 16; i += 128) {
    int t = i >> 4, n0 = (i & 15) * 4;
    v4f cv = *(const v4f*)(xc + (size_t)(T0 + t) * CONVD + DIN + DSTATE + n0);
    v4bf cc;
#pragma unroll
    for (int u = 0; u < 4; ++u) cc[u] = (__bf16)cv[u];
    *(v4bf*)(sC + t * 64 + n0) = cc;
  }
  for (int i = tid; i < HD * 16; i += 128) {
    int p = i >> 4, n0 = (i & 15) * 4;
    v4f hv = *(const v4f*)(Hin + (size_t)(c * NH + h) * 2048 + p * 64 + n0);
    v4bf hh;
#pragma unroll
    for (int u = 0; u < 4; ++u) hh[u] = (__bf16)hv[u];
    *(v4bf*)(sH + p * 64 + n0) = hh;
  }
  __syncthreads();

#pragma unroll
  for (int i = 0; i < 2; ++i) {
    int tile = wave * 2 + i;      // 8 tiles (4 x 2)
    int tm = tile >> 1, tn = tile & 1;
    v8f acc = {};
#pragma unroll
    for (int k0 = 0; k0 < 64; k0 += 32) {
      v16bf a = load_a_frag(sC + tm * 16 * 64 + k0, 64, lane);
      v16bf b = load_b_frag(sH + tn * 16 * 64 + k0, 64, lane);
      acc = wmma_bf16(a, b, acc);
    }
    int r0 = tm * 16 + ((lane >> 4) << 3);
    int p  = tn * 16 + (lane & 15);
#pragma unroll
    for (int r = 0; r < 8; ++r) {
      int t = T0 + r0 + r;
      size_t oi = (size_t)t * DIN + h * HD + p;
      ysv[oi] += acc[r] * Pglob[(size_t)t * NH + h];
    }
  }
}

// ---------------------------------------------------------------------------
// y = (ys + D*xs) * silu(z);  RMSNorm over D_INNER;  store bf16 (float4 path)
// ---------------------------------------------------------------------------
__global__ void k_gate_norm(const float* __restrict__ ys,
                            const float* __restrict__ xc,
                            const float* __restrict__ zx,
                            const float* __restrict__ Dp,
                            const float* __restrict__ norm_w,
                            __bf16* __restrict__ yn) {
  int t = blockIdx.x;
  int j = threadIdx.x;            // 256 threads, 4 consecutive elements each
  int i0 = j * 4;
  __shared__ float red[256];

  v4f yv = *(const v4f*)(ys + (size_t)t * DIN + i0);
  v4f xv = *(const v4f*)(xc + (size_t)t * CONVD + i0);
  v4f zv = *(const v4f*)(zx + (size_t)t * DIP + i0);
  float dcoef = Dp[i0 >> 5];

  float vals[4];
  float ss = 0.f;
#pragma unroll
  for (int u = 0; u < 4; ++u) {
    float y = yv[u] + dcoef * xv[u];
    float z = zv[u];
    y *= z / (1.f + __expf(-z));
    vals[u] = y;
    ss += y * y;
  }
  red[j] = ss;
  __syncthreads();
  for (int s = 128; s > 0; s >>= 1) {
    if (j < s) red[j] += red[j + s];
    __syncthreads();
  }
  float inv = rsqrtf(red[0] * (1.f / DIN) + 1e-5f);

  v4f nw = *(const v4f*)(norm_w + i0);
  v4bf o;
#pragma unroll
  for (int u = 0; u < 4; ++u) o[u] = (__bf16)(vals[u] * inv * nw[u]);
  *(v4bf*)(yn + (size_t)t * DIN + i0) = o;
}

// ---------------------------------------------------------------------------
// scores[t,s] = (t>=s) ? (q[t].k[s]) / 8 : -inf.
// Wave handles a 16x64 strip: q fragments loaded once, reused for 4 k tiles.
// Fully-masked tiles: two b128 -inf stores per lane.
// ---------------------------------------------------------------------------
__global__ void k_scores(const __bf16* __restrict__ q,
                         const __bf16* __restrict__ k,
                         float* __restrict__ out) {
  int lane = threadIdx.x & 31;
  int wave = threadIdx.x >> 5;
  int nt0 = (blockIdx.x * 4 + wave) * 4;   // first s tile
  int mt  = blockIdx.y;                    // t tile

  int r0 = mt * 16 + ((lane >> 4) << 3);
  const __bf16* qb = q + (size_t)mt * 16 * IDXD;
  v16bf a0 = load_a_frag(qb + 0,  IDXD, lane);
  v16bf a1 = load_a_frag(qb + 32, IDXD, lane);
  const float NEGINF = -__builtin_inff();
  const float scale = 0.125f;              // IDX_DIM^-0.5

#pragma unroll
  for (int j = 0; j < 4; ++j) {
    int nt = nt0 + j;
    if (mt * 16 + 15 < nt * 16) {          // tile strictly above diagonal
      int row  = mt * 16 + (lane >> 1);    // 16 rows x 2 lanes
      int col  = nt * 16 + (lane & 1) * 8; // 8 cols per lane -> 2x b128
      v4f m4 = {NEGINF, NEGINF, NEGINF, NEGINF};
      float* p = out + (size_t)row * LQ + col;
      *(v4f*)(p)     = m4;
      *(v4f*)(p + 4) = m4;
      continue;
    }
    const __bf16* kb = k + (size_t)nt * 16 * IDXD;
    v8f acc = {};
    acc = wmma_bf16(a0, load_b_frag(kb + 0,  IDXD, lane), acc);
    acc = wmma_bf16(a1, load_b_frag(kb + 32, IDXD, lane), acc);
    int c = nt * 16 + (lane & 15);
#pragma unroll
    for (int r = 0; r < 8; ++r) {
      int row = r0 + r;
      out[(size_t)row * LQ + c] = (row >= c) ? acc[r] * scale : NEGINF;
    }
  }
}

// ---------------------------------------------------------------------------
extern "C" void kernel_launch(void* const* d_in, const int* in_sizes, int n_in,
                              void* d_out, int out_size, void* d_ws, size_t ws_size,
                              hipStream_t stream) {
  const float* x       = (const float*)d_in[0];
  const float* W_in    = (const float*)d_in[1];
  const float* conv_w  = (const float*)d_in[2];
  const float* conv_b  = (const float*)d_in[3];
  const float* dt_bias = (const float*)d_in[4];
  const float* A_log   = (const float*)d_in[5];
  const float* Dp      = (const float*)d_in[6];
  const float* norm_w  = (const float*)d_in[7];
  const float* W_out   = (const float*)d_in[8];
  const float* W_q     = (const float*)d_in[9];
  const float* W_k     = (const float*)d_in[10];
  float* out = (float*)d_out;

  char* ws = (char*)d_ws;
  size_t off = 0;
  auto alloc = [&](size_t bytes) -> void* {
    off = (off + 255) & ~(size_t)255;
    void* p = ws + off;
    off += bytes;
    return p;
  };

  __bf16* x_bf    = (__bf16*)alloc((size_t)LQ * DM * 2);
  __bf16* Win_bf  = (__bf16*)alloc((size_t)DIP * DM * 2);
  __bf16* Wout_bf = (__bf16*)alloc((size_t)DM * DIN * 2);
  __bf16* Wq_bf   = (__bf16*)alloc((size_t)IDXD * DM * 2);
  __bf16* Wk_bf   = (__bf16*)alloc((size_t)IDXD * DM * 2);
  float*  zx      = (float*) alloc((size_t)LQ * DIP * 4);
  float*  xc      = (float*) alloc((size_t)LQ * CONVD * 4);
  float*  dtp     = (float*) alloc((size_t)LQ * NH * 4);
  float*  ldA     = (float*) alloc((size_t)LQ * NH * 4);
  float*  ysv     = (float*) alloc((size_t)LQ * DIN * 4);
  float*  Uglob   = (float*) alloc((size_t)NC * NH * HD * 64 * 4);
  float*  Hin     = (float*) alloc((size_t)NC * NH * HD * 64 * 4);
  float*  Pglob   = (float*) alloc((size_t)LQ * NH * 4);
  float*  Ptot    = (float*) alloc((size_t)NC * NH * 4);
  __bf16* yn_bf   = (__bf16*)alloc((size_t)LQ * DIN * 2);
  __bf16* xm_bf   = (__bf16*)alloc((size_t)LQ * DM * 2);
  __bf16* q_bf    = (__bf16*)alloc((size_t)LQ * IDXD * 2);
  __bf16* k_bf    = (__bf16*)alloc((size_t)LQ * IDXD * 2);

  auto cvt = [&](const float* src, __bf16* dst, int n) {
    int n4 = n / 4;
    k_f32_to_bf16<<<(n4 + 255) / 256, 256, 0, stream>>>(src, dst, n4);
  };
  cvt(x,     x_bf,    LQ * DM);
  cvt(W_in,  Win_bf,  DIP * DM);
  cvt(W_out, Wout_bf, DM * DIN);
  cvt(W_q,   Wq_bf,   IDXD * DM);
  cvt(W_k,   Wk_bf,   IDXD * DM);

  // in-projection: zx[L,DIP] = x @ W_in^T
  {
    int ntiles = DIP / 16;                      // 138
    k_gemm_wmma<false><<<dim3((ntiles + 15) / 16, LQ / 16), 128, 0, stream>>>(
        x_bf, Win_bf, zx, nullptr, LQ, DIP, DM);
  }

  // depthwise conv + silu
  k_conv_silu<<<(LQ * (CONVD / 4) + 255) / 256, 256, 0, stream>>>(zx, conv_w, conv_b, xc);

  // dt / log(dA)
  k_dt<<<(LQ * NH + 255) / 256, 256, 0, stream>>>(zx, dt_bias, A_log, dtp, ldA);

  // chunked SSD scan
  k_ssd_chunk<<<dim3(NC, NH), 128, 0, stream>>>(xc, dtp, ldA, ysv, Uglob, Pglob, Ptot);
  k_chunk_state<<<NH, 1024, 0, stream>>>(Uglob, Ptot, Hin);
  k_ssd_inter<<<dim3(NC, NH), 128, 0, stream>>>(xc, Hin, Pglob, ysv);

  // gate + RMSNorm -> bf16
  k_gate_norm<<<LQ, 256, 0, stream>>>(ysv, xc, zx, Dp, norm_w, yn_bf);

  // out-projection: xm[L,DM] = yn @ W_out^T (bf16 out)
  k_gemm_wmma<true><<<dim3((DM / 16) / 16, LQ / 16), 128, 0, stream>>>(
      yn_bf, Wout_bf, nullptr, xm_bf, LQ, DM, DIN);

  // q = xm @ W_q^T ; k = x @ W_k^T
  k_gemm_wmma<true><<<dim3(1, LQ / 16), 128, 0, stream>>>(
      xm_bf, Wq_bf, nullptr, q_bf, LQ, IDXD, DM);
  k_gemm_wmma<true><<<dim3(1, LQ / 16), 128, 0, stream>>>(
      x_bf, Wk_bf, nullptr, k_bf, LQ, IDXD, DM);

  // causal score matrix
  k_scores<<<dim3(LQ / 16 / 16, LQ / 16), 128, 0, stream>>>(q_bf, k_bf, out);
}
